// ScaledDotProductAttention_7138235646088
// MI455X (gfx1250) — matrix operand order repaired and checked by reading, not compiled, not run on hardware
//
#include <hip/hip_runtime.h>
#include <hip/hip_bf16.h>
#include <math.h>

typedef __attribute__((ext_vector_type(16))) _Float16 v16h;
typedef __attribute__((ext_vector_type(8)))  _Float16 v8h;
typedef __attribute__((ext_vector_type(4)))  _Float16 v4h;
typedef __attribute__((ext_vector_type(8)))  float    v8f;

#define B_   8
#define SQ_  2048
#define SK_  2048
#define D_   128
#define NEG_ (-1.0e9f)
#define INV_TEMPER_ 0.08838834764831845f   /* 1/sqrt(128) */

// Main-kernel LDS float layout (dynamic):
//   sS   : 8 waves * 16 rows * 256 cols   = 32768 floats (S, then P)
//   sO   : 16 * 128                       =  2048 floats (output accumulation)
//   sRed : 16 rows * 16 partials          =   256 floats
//   sMax : 16, sInv : 16                  =    32 floats
#define SMEM_FLOATS_ (32768 + 2048 + 256 + 32)

// ---------------------------------------------------------------------------
// Pre-pass 1: K f32 [B,Sk,D] -> f16 [B,Sk,D] (elementwise, coalesced)
// ---------------------------------------------------------------------------
__global__ __launch_bounds__(256)
void cvt_k_kernel(const float* __restrict__ k, _Float16* __restrict__ kf16)
{
    const size_t base = ((size_t)blockIdx.x * 256 + threadIdx.x) * 4;
    const float4 f = *(const float4*)(k + base);
    v4h h;
    h[0] = (_Float16)f.x; h[1] = (_Float16)f.y;
    h[2] = (_Float16)f.z; h[3] = (_Float16)f.w;
    *(v4h*)(kf16 + base) = h;
}

// ---------------------------------------------------------------------------
// Pre-pass 2: V f32 [B,Sk,D] -> Vt f16 [B,D,Sk] (LDS-tiled transpose)
// grid = (Sk/64, D/64, B), block = 256
// ---------------------------------------------------------------------------
__global__ __launch_bounds__(256)
void transpose_v_kernel(const float* __restrict__ v, _Float16* __restrict__ vt)
{
    __shared__ _Float16 sT[64 * 72];   // [d_local][k_local], stride 72 (16B-aligned rows)

    const int tid = threadIdx.x;
    const int k0  = blockIdx.x * 64;
    const int d0  = blockIdx.y * 64;
    const int b   = blockIdx.z;

    // load 64x64 f32 tile coalesced, scatter transposed into LDS as f16
    {
        const int r  = tid >> 2;           // k row in tile, 0..63
        const int qd = (tid & 3) * 16;     // starting d col in tile
        const float* src = v + ((size_t)b * SK_ + (k0 + r)) * D_ + d0 + qd;
        #pragma unroll
        for (int i = 0; i < 16; i += 4) {
            const float4 f = *(const float4*)(src + i);
            sT[(qd + i + 0) * 72 + r] = (_Float16)f.x;
            sT[(qd + i + 1) * 72 + r] = (_Float16)f.y;
            sT[(qd + i + 2) * 72 + r] = (_Float16)f.z;
            sT[(qd + i + 3) * 72 + r] = (_Float16)f.w;
        }
    }
    __syncthreads();

    // write rows of Vt coalesced (each thread: 16 contiguous f16 = 32B)
    {
        const int dl = tid >> 2;           // d row in tile
        const int qk = (tid & 3) * 16;     // starting k col in tile
        _Float16* dst = vt + ((size_t)b * D_ + (d0 + dl)) * SK_ + k0 + qk;
        const v8h* s = (const v8h*)&sT[dl * 72 + qk];
        *(v8h*)(dst)     = s[0];
        *(v8h*)(dst + 8) = s[1];
    }
}

// ---------------------------------------------------------------------------
// Main attention kernel: one workgroup (8 waves) per (batch, 16-row q-tile)
// ---------------------------------------------------------------------------
__global__ __launch_bounds__(256)
void ScaledDotProductAttention_7138235646088_kernel(
    const float* __restrict__ q, const _Float16* __restrict__ kf16,
    const _Float16* __restrict__ vt, const int* __restrict__ mask,
    float* __restrict__ out, float* __restrict__ attn)
{
    extern __shared__ char smem_raw[];
    float* sS   = (float*)smem_raw;          // [ (w*16+m)*256 + c ]
    float* sO   = sS + 32768;                // [ m*128 + d ]
    float* sRed = sO + 2048;                 // [ row*16 + seg ]
    float* sMax = sRed + 256;                // [ row ]
    float* sInv = sMax + 16;                 // [ row ]

    const int tid  = threadIdx.x;
    const int wave = tid >> 5;
    const int lane = tid & 31;
    const int lm   = lane & 15;     // M (A/C) or N (B/C) index within half
    const int lh   = lane >> 4;     // which half of the wave

    const int b  = blockIdx.y;
    const int q0 = blockIdx.x * 16;
    const size_t qRowBase = (size_t)b * SQ_ + q0;

    // zero the shared output accumulator (consumed after later barriers)
    #pragma unroll
    for (int i = 0; i < 8; ++i) sO[tid + 256 * i] = 0.0f;

    // ---------------- Phase A: S = Q K^T * invTemper + NEG*(1-mask) --------
    // A 16x32 as loaded here (contiguous k order): lane(m=lm, h=lh); elem e -> d = 32c+16h+e
    v16h qf[4];
    {
        const float* qrow = q + (qRowBase + lm) * D_;
        #pragma unroll
        for (int c = 0; c < 4; ++c) {
            const float* r0 = qrow + 32 * c + 16 * lh;
            const float* r1 = qrow + 32 * c + 16 * lh + 8;
            #pragma unroll
            for (int e = 0; e < 8; ++e) {
                qf[c][e]     = (_Float16)r0[e];
                qf[c][e + 8] = (_Float16)r1[e];
            }
        }
    }

    const int colBase = wave * 256;          // this wave's 256-col chunk of Sk
    #pragma unroll 1
    for (int t = 0; t < 16; ++t) {           // 16 tiles of 16 columns
        const int n0 = colBase + 16 * t;
        const int n  = n0 + lm;              // B-matrix: lane -> column of S
        v8f acc = {};
        #pragma unroll
        for (int c = 0; c < 4; ++c) {
            // B 32x16 as loaded here (contiguous k order, same as A): lane(n, h); elem e -> d = 32c + 16h + e
            // Kf16 row is contiguous: single aligned 32B v16h load
            const v16h kf = *(const v16h*)(kf16 +
                ((size_t)b * SK_ + n) * D_ + 32 * c + 16 * lh);
            acc = __builtin_amdgcn_wmma_f32_16x16x32_f16(
                false, qf[c], false, kf, (short)0, acc, false, false);
        }
        // scale + additive mask, store to LDS. C layout: VGPR r ->
        // (m = r + 8*lh, col = n0 + lm)
        #pragma unroll
        for (int r = 0; r < 8; ++r) {
            const int m = r + 8 * lh;
            const int col = n0 + lm;
            const int mk = mask[(qRowBase + m) * SK_ + col];
            float s = acc[r] * INV_TEMPER_ + NEG_ * (1.0f - (float)mk);
            sS[(wave * 16 + m) * 256 + (16 * t + lm)] = s;
        }
    }
    __syncthreads();

    // ---------------- Phase B: row-wise softmax statistics -----------------
    // Strip mapping: row = tid>>4, seg = tid&15, cols = seg + 16*j (coalesced)
    const int row = tid >> 4;
    const int seg = tid & 15;
    {
        float mx = -INFINITY;
        #pragma unroll 4
        for (int j = 0; j < 128; ++j) {
            const int col = seg + 16 * j;
            mx = fmaxf(mx, sS[((col >> 8) * 16 + row) * 256 + (col & 255)]);
        }
        sRed[row * 16 + seg] = mx;
    }
    __syncthreads();
    if (tid < 16) {
        float mx = -INFINITY;
        #pragma unroll
        for (int s2 = 0; s2 < 16; ++s2) mx = fmaxf(mx, sRed[tid * 16 + s2]);
        sMax[tid] = mx;
    }
    __syncthreads();
    {
        const float mrow = sMax[row];
        float sum = 0.0f;
        #pragma unroll 4
        for (int j = 0; j < 128; ++j) {
            const int col = seg + 16 * j;
            sum += __expf(sS[((col >> 8) * 16 + row) * 256 + (col & 255)] - mrow);
        }
        sRed[row * 16 + seg] = sum;
    }
    __syncthreads();
    if (tid < 16) {
        float sum = 0.0f;
        #pragma unroll
        for (int s2 = 0; s2 < 16; ++s2) sum += sRed[tid * 16 + s2];
        sInv[tid] = 1.0f / sum;
    }
    __syncthreads();

    // ---------------- Phase C: P = softmax(S); write attn; P back to LDS ---
    {
        const float mrow = sMax[row];
        const float inv  = sInv[row];
        float* arow = attn + (qRowBase + row) * SK_;
        #pragma unroll 4
        for (int j = 0; j < 128; ++j) {
            const int col = seg + 16 * j;
            const int li = ((col >> 8) * 16 + row) * 256 + (col & 255);
            const float p = __expf(sS[li] - mrow) * inv;
            arow[col] = p;          // coalesced 64B segments per 16-thread row
            sS[li] = p;
        }
    }
    __syncthreads();

    // ---------------- Phase D: O = P V  (per-wave k-chunk, WMMA) -----------
    v8f oacc[8];
    #pragma unroll
    for (int dt = 0; dt < 8; ++dt) oacc[dt] = (v8f){};

    #pragma unroll 1
    for (int kc = 0; kc < 8; ++kc) {         // 8 chunks of K=32 in wave's 256
        const int kloc  = kc * 32;
        const int kglob = colBase + kloc;
        // A-fragment of P from LDS: lane(m=lm,h=lh);
        // elem e -> col = kloc+16h+e
        v16h pa;
        {
            const float* p0 = &sS[(wave * 16 + lm) * 256 + kloc + 16 * lh];
            const float* p1 = p0 + 8;
            #pragma unroll
            for (int e = 0; e < 8; ++e) {
                pa[e]     = (_Float16)p0[e];
                pa[e + 8] = (_Float16)p1[e];
            }
        }
        #pragma unroll 1
        for (int dt = 0; dt < 8; ++dt) {     // 8 output column tiles of D=128
            const int n = dt * 16 + lm;      // output column (d index)
            // B-fragment of V from Vt[b][d][k]: 16 contiguous k at fixed d:
            // single aligned 32B v16h load
            const v16h vb = *(const v16h*)(vt +
                ((size_t)b * D_ + n) * SK_ + kglob + 16 * lh);
            oacc[dt] = __builtin_amdgcn_wmma_f32_16x16x32_f16(
                false, pa, false, vb, (short)0, oacc[dt], false, false);
        }
    }
    // reduce partial O across the 8 waves via LDS float atomics
    #pragma unroll
    for (int dt = 0; dt < 8; ++dt) {
        #pragma unroll
        for (int r = 0; r < 8; ++r) {
            const int m = r + 8 * lh;
            const int d = dt * 16 + lm;
            atomicAdd(&sO[m * 128 + d], oacc[dt][r]);
        }
    }
    __syncthreads();

    // ---------------- store O (16 x 128), coalesced ------------------------
    #pragma unroll
    for (int i = 0; i < 8; ++i) {
        const int idx = tid + 256 * i;       // 0..2047
        const int m = idx >> 7;
        const int d = idx & 127;
        out[(qRowBase + m) * D_ + d] = sO[idx];
    }
}

extern "C" void kernel_launch(void* const* d_in, const int* in_sizes, int n_in,
                              void* d_out, int out_size, void* d_ws, size_t ws_size,
                              hipStream_t stream) {
    (void)in_sizes; (void)n_in; (void)out_size; (void)ws_size;
    const float* q    = (const float*)d_in[0];
    const float* k    = (const float*)d_in[1];
    const float* v    = (const float*)d_in[2];
    const int*   mask = (const int*)d_in[3];

    float* out  = (float*)d_out;                              // [B,Sq,D]
    float* attn = out + (size_t)B_ * SQ_ * D_;                // [B,Sq,Sk]

    // workspace: Kf16 [B,Sk,D] (4MB) + Vt f16 [B,D,Sk] (4MB)
    _Float16* kf16 = (_Float16*)d_ws;
    _Float16* vtp  = kf16 + (size_t)B_ * SK_ * D_;

    // pre-pass 1: K -> f16
    {
        const size_t nElem = (size_t)B_ * SK_ * D_;
        dim3 grid((unsigned)(nElem / (256 * 4)));
        cvt_k_kernel<<<grid, 256, 0, stream>>>(k, kf16);
    }
    // pre-pass 2: V -> f16 transposed [B,D,Sk]
    {
        dim3 grid(SK_ / 64, D_ / 64, B_);
        transpose_v_kernel<<<grid, 256, 0, stream>>>(v, vtp);
    }

    const size_t smemBytes = (size_t)SMEM_FLOATS_ * sizeof(float); // ~137 KB
    static_assert(SMEM_FLOATS_ * 4 <= 320 * 1024, "fits WGP LDS");

    hipFuncSetAttribute(
        (const void*)ScaledDotProductAttention_7138235646088_kernel,
        hipFuncAttributeMaxDynamicSharedMemorySize, (int)smemBytes);

    dim3 grid(SQ_ / 16, B_);   // 128 x 8 workgroups
    dim3 block(256);           // 8 waves (wave32)
    ScaledDotProductAttention_7138235646088_kernel<<<grid, block, smemBytes, stream>>>(
        q, kf16, vtp, mask, out, attn);
}
